// TGraphMultiHeadAttention_10574209483496
// MI455X (gfx1250) — matrix operand order repaired and checked by reading, not compiled, not run on hardware
//
#include <hip/hip_runtime.h>

// ---------------------------------------------------------------------------
// TGraphMultiHeadAttention on MI455X (gfx1250, wave32, WMMA bf16 16x16x32)
//   N=4096 nodes, D=256, F=128, H=8 heads, DQ=32
// Pipeline (all bf16 WMMA, f32 accumulate):
//   1) h_x  = relu(x @ Wself + bself)            -> bf16 row-major
//   2) supn = (x @ Wn) * t                        -> bf16 K-major (supT)
//   3) supc = (x @ Wc) * t                        -> bf16 K-major (supT)
//   4) h_n  = relu(adj @ supn + bn)               -> bf16 row-major
//   5) h_c  = relu((adj+I) @ supc + bc)           -> bf16 row-major
//   6) q = (h_x@Wq+bq)/sqrt(DQ), k = h_n@Wk+bk;  v = h_c@Wv+bv (K-major vT)
//   7) flash attention, transposed-S form: sT = K@qT (lane = query), per-lane
//      online softmax, P routed via v_permlanex16 (no LDS), oT = vT@P
//   8) out  = o @ Wo + bo (f32)
// ---------------------------------------------------------------------------

constexpr int kN  = 4096;
constexpr int kD  = 256;
constexpr int kF  = 128;
constexpr int kH  = 8;
constexpr int kDQ = 32;

typedef __attribute__((ext_vector_type(16))) __bf16 bf16x16;
typedef __attribute__((ext_vector_type(8)))  float  v8f;

union ABFrag {
  bf16x16  v;
  uint4    q[2];
  unsigned u[8];
};

__device__ __forceinline__ unsigned short f2bf(float f) {
  unsigned int u = __float_as_uint(f);
  u += 0x7FFFu + ((u >> 16) & 1u);   // round-to-nearest-even
  return (unsigned short)(u >> 16);
}
__device__ __forceinline__ unsigned packbf2(float lo, float hi) {
  return (unsigned)f2bf(lo) | ((unsigned)f2bf(hi) << 16);
}

__device__ __forceinline__ v8f zero8() {
  v8f z;
#pragma unroll
  for (int i = 0; i < 8; ++i) z[i] = 0.0f;
  return z;
}

__device__ __forceinline__ v8f wmma_bf16(bf16x16 a, bf16x16 b, v8f c) {
  // D = A(16x32) * B(32x16) + C, f32 accumulate
  return __builtin_amdgcn_wmma_f32_16x16x32_bf16(false, a, false, b,
                                                 (short)0, c, false, false);
}

// ---- cross-half (lane ^ 16) exchange via v_permlanex16, identity mapping ----
__device__ __forceinline__ unsigned xhalf_u(unsigned v) {
  return (unsigned)__builtin_amdgcn_permlanex16(
      (int)v, (int)v, 0x76543210, (int)0xFEDCBA98u, false, false);
}
__device__ __forceinline__ float xhalf_f(float v) {
  return __int_as_float(__builtin_amdgcn_permlanex16(
      __float_as_int(v), __float_as_int(v), 0x76543210, (int)0xFEDCBA98u,
      false, false));
}

// ---------------------------------------------------------------------------
// Generic bf16-WMMA GEMM:  out[h] = EPI( A[h] @ B[h] + bias[h] )
//   Tile: M=128 (8 waves x 16 rows), Ncols = NT (full), K looped in chunks of 32.
//   ABF   : A is bf16 row-major in global (fragments loaded directly)
//           else A is f32 row-major, staged+converted into LDS (double buffered)
//   IDENT : add identity matrix to A (adj + I), f32 path only
//   BPT   : B is bf16 K-major [ncol][K] in global (supT) -> staged into LDS
//           with GLOBAL_LOAD_ASYNC_TO_LDS_B128 (ASYNCcnt-tracked); else B is
//           f32 row-major [K][ncol] (weights), transpose-staged into LDS
//   EPI   : 0=none, 1=relu, 2=scale rows by t[row], 3=scale by cscale
//   OMODE : 0 = bf16 row-major, 1 = bf16 K-major (transposed), 2 = f32 row-major
// ---------------------------------------------------------------------------
template <int NT, bool ABF, bool IDENT, bool BPT, int EPI, bool BIAS, int OMODE>
__global__ __launch_bounds__(256) void gemm_wmma_kernel(
    const float* __restrict__ Af, const unsigned short* __restrict__ Ab,
    long ahs, int lda,
    const float* __restrict__ Bf, const unsigned short* __restrict__ Bp,
    long bhs, int ldb,
    const float* __restrict__ bias, int biashs,
    const float* __restrict__ tvec, float cscale,
    float* __restrict__ outf, unsigned short* __restrict__ outb,
    long ohs, int ldo, int K) {
  constexpr int MT   = 128;
  constexpr int NSUB = NT / 16;

  __shared__ unsigned short As[2][ABF ? 32 : MT * 32];
  __shared__ unsigned short Bs[2][NT * 32];

  const int h     = blockIdx.y;
  const int mbase = blockIdx.x * MT;
  const int tid   = threadIdx.x;
  const int lane  = tid & 31;
  const int w     = tid >> 5;
  const int hi    = lane >> 4;
  const int nl    = lane & 15;

  v8f acc[NSUB];
#pragma unroll
  for (int nt = 0; nt < NSUB; ++nt) acc[nt] = zero8();

  const int KC = K / 32;

  auto stage = [&](int buf, int kc) {
    if constexpr (!ABF) {
      // 128x32 f32 -> bf16 LDS; each thread converts 16 consecutive elems
      const int r  = tid >> 1;
      const int c0 = (tid & 1) * 16;
      const float* src = Af + (size_t)(mbase + r) * lda + kc * 32 + c0;
      unsigned short* dst = &As[buf][r * 32 + c0];
      if (kc + 2 < KC) __builtin_prefetch(src + 64, 0, 0);  // global_prefetch
#pragma unroll
      for (int i = 0; i < 16; ++i) {
        float vx = src[i];
        if constexpr (IDENT) {
          if (mbase + r == kc * 32 + c0 + i) vx += 1.0f;
        }
        dst[i] = f2bf(vx);
      }
    }
    if constexpr (BPT) {
      // B already bf16 K-major: async 16B copies global -> LDS (ASYNCcnt)
#pragma unroll
      for (int i = 0; i < (NT * 4 + 255) / 256; ++i) {
        int idx = tid + i * 256;
        if (idx < NT * 4) {
          int r = idx >> 2, q4 = idx & 3;
          const unsigned short* src =
              Bp + (size_t)h * bhs + (size_t)r * ldb + kc * 32 + q4 * 8;
          unsigned ldsoff =
              (unsigned)(uintptr_t)&Bs[buf][r * 32 + q4 * 8];  // addr[31:0]
          asm volatile("global_load_async_to_lds_b128 %0, %1, off"
                       :: "v"(ldsoff), "v"(src)
                       : "memory");
        }
      }
    } else {
      // f32 weights [K][NT] -> transposed bf16 LDS [NT][32]
#pragma unroll
      for (int i = 0; i < (NT * 32) / 256; ++i) {
        int e = tid + i * 256;
        int f = e % NT, d = e / NT;
        float vx = Bf[(size_t)h * bhs + (size_t)(kc * 32 + d) * ldb + f];
        Bs[buf][f * 32 + d] = f2bf(vx);
      }
    }
  };

  stage(0, 0);
  for (int kc = 0; kc < KC; ++kc) {
    if constexpr (BPT)  // our async stage from previous iter must have landed
      asm volatile("s_wait_asynccnt 0" ::: "memory");
    __syncthreads();
    const int buf = kc & 1;
    if (kc + 1 < KC) stage(buf ^ 1, kc + 1);

    // A fragment: lane row = w*16 + nl; k chunks at 8*hi and 16+8*hi
    ABFrag a;
    if constexpr (ABF) {
      const unsigned short* ar =
          Ab + (size_t)h * ahs + (size_t)(mbase + w * 16 + nl) * lda + kc * 32;
      a.q[0] = *(const uint4*)(ar + hi * 8);
      a.q[1] = *(const uint4*)(ar + 16 + hi * 8);
    } else {
      const unsigned short* ar = &As[buf][(w * 16 + nl) * 32];
      a.q[0] = *(const uint4*)(ar + hi * 8);
      a.q[1] = *(const uint4*)(ar + 16 + hi * 8);
    }
#pragma unroll
    for (int nt = 0; nt < NSUB; ++nt) {
      // B fragment: column = nt*16 + nl, k chunks at 8*hi and 16+8*hi
      ABFrag b;
      const unsigned short* br = &Bs[buf][(nt * 16 + nl) * 32 + hi * 8];
      b.q[0] = *(const uint4*)br;
      b.q[1] = *(const uint4*)(br + 16);
      acc[nt] = wmma_bf16(a.v, b.v, acc[nt]);
    }
  }

  // Epilogue: D layout = lane n = nl (+16*hi selects m-half), m = 8*hi + j
#pragma unroll
  for (int nt = 0; nt < NSUB; ++nt) {
    const int col = nt * 16 + nl;
    float bv = 0.0f;
    if constexpr (BIAS) bv = bias[(size_t)h * biashs + col];
    if constexpr (OMODE == 1) {
      alignas(16) unsigned short tmp[8];
#pragma unroll
      for (int j = 0; j < 8; ++j) {
        int row  = mbase + w * 16 + hi * 8 + j;
        float vx = acc[nt][j] + bv;
        if constexpr (EPI == 1) vx = fmaxf(vx, 0.0f);
        if constexpr (EPI == 2) vx *= tvec[row];
        if constexpr (EPI == 3) vx *= cscale;
        tmp[j] = f2bf(vx);
      }
      *(uint4*)&outb[(size_t)h * ohs + (size_t)col * ldo + mbase + w * 16 +
                     hi * 8] = *(const uint4*)tmp;
    } else {
#pragma unroll
      for (int j = 0; j < 8; ++j) {
        int row  = mbase + w * 16 + hi * 8 + j;
        float vx = acc[nt][j] + bv;
        if constexpr (EPI == 1) vx = fmaxf(vx, 0.0f);
        if constexpr (EPI == 2) vx *= tvec[row];
        if constexpr (EPI == 3) vx *= cscale;
        if constexpr (OMODE == 0)
          outb[(size_t)h * ohs + (size_t)row * ldo + col] = f2bf(vx);
        if constexpr (OMODE == 2)
          outf[(size_t)h * ohs + (size_t)row * ldo + col] = vx;
      }
    }
  }
}

// ---------------------------------------------------------------------------
// Flash attention, transposed-S form. Per (head, 128-query block), 8 waves x
// 16 queries. For each 32-key chunk:
//   sT(16 keys x 16 queries) = A(k rows) @ B(q rows)  -> lane = query, m = key
//   per-lane (per-query) online softmax; single cross-half permlanex16 combine
//   P packed to bf16 in registers, routed to B-fragment with permlanex16
//   oT(dq x queries) += A(vT rows) @ P
//   q  : bf16 [H][N][DQ], pre-scaled by 1/sqrt(DQ)
//   k  : bf16 [H][N][DQ]
//   vT : bf16 [H][DQ][N]
//   oBuf f32 [N][H*DQ]
// ---------------------------------------------------------------------------
__global__ __launch_bounds__(256) void attn_kernel(
    const unsigned short* __restrict__ qb, const unsigned short* __restrict__ kb,
    const unsigned short* __restrict__ vT, float* __restrict__ oBuf) {
  const int h    = blockIdx.y;
  const int tid  = threadIdx.x;
  const int lane = tid & 31;
  const int w    = tid >> 5;
  const int hi   = lane >> 4;
  const int nl   = lane & 15;
  const int qrow = blockIdx.x * 128 + w * 16 + nl;  // this lane's query

  // B fragment = q rows (column = query nl, K = dq chunks 8*hi, 16+8*hi); held all loop
  const unsigned short* qp = qb + ((size_t)h * kN + qrow) * kDQ + hi * 8;
  ABFrag bq;
  bq.q[0] = *(const uint4*)qp;
  bq.q[1] = *(const uint4*)(qp + 16);

  float mrun = -1e30f, lrun = 0.0f;     // per-query scalars (lane-owned)
  v8f ot0 = zero8(), ot1 = zero8();     // oT accum: dq = 8*hi+j / 16+8*hi+j

  for (int kbase = 0; kbase < kN; kbase += 32) {
    // A fragments = k rows (16 keys x 32 dq), two key subtiles
    const unsigned short* kp0 = kb + ((size_t)h * kN + kbase + nl) * kDQ;
    const unsigned short* kp1 = kp0 + 16 * kDQ;
    ABFrag ak0, ak1;
    ak0.q[0] = *(const uint4*)(kp0 + hi * 8);
    ak0.q[1] = *(const uint4*)(kp0 + 16 + hi * 8);
    ak1.q[0] = *(const uint4*)(kp1 + hi * 8);
    ak1.q[1] = *(const uint4*)(kp1 + 16 + hi * 8);

    // sT: lane = query nl; st0[j] = key kbase+8*hi+j ; st1[j] = +16
    v8f st0 = wmma_bf16(ak0.v, bq.v, zero8());
    v8f st1 = wmma_bf16(ak1.v, bq.v, zero8());

    // per-lane max over this lane's 16 scores (tree), then cross-half combine
    float red[8];
#pragma unroll
    for (int j = 0; j < 8; ++j) red[j] = fmaxf(st0[j], st1[j]);
#pragma unroll
    for (int s = 4; s >= 1; s >>= 1)
#pragma unroll
      for (int i = 0; i < s; ++i) red[i] = fmaxf(red[i], red[i + s]);
    float mx = fmaxf(red[0], xhalf_f(red[0]));  // full 32-key chunk max

    float mn   = fmaxf(mrun, mx);
    float corr = __expf(mrun - mn);
    float p0[8], p1[8];
#pragma unroll
    for (int j = 0; j < 8; ++j) {
      p0[j] = __expf(st0[j] - mn);
      p1[j] = __expf(st1[j] - mn);
    }
    float sr[8];
#pragma unroll
    for (int j = 0; j < 8; ++j) sr[j] = p0[j] + p1[j];
#pragma unroll
    for (int s = 4; s >= 1; s >>= 1)
#pragma unroll
      for (int i = 0; i < s; ++i) sr[i] += sr[i + s];
    float rs = sr[0] + xhalf_f(sr[0]);

    lrun = lrun * corr + rs;
    mrun = mn;
#pragma unroll
    for (int j = 0; j < 8; ++j) { ot0[j] *= corr; ot1[j] *= corr; }

    // Build P B-fragment (32 keys x 16 queries): column = query nl,
    // elements K = keys 16*hi..+15.  Own lane holds keys {8hi..8hi+7} (st0)
    // and {16+8hi..+7} (st1); fetch the other 8-run from lane^16.
    unsigned pk0[4], pk1[4], x0[4], x1[4];
#pragma unroll
    for (int e = 0; e < 4; ++e) {
      pk0[e] = packbf2(p0[2 * e], p0[2 * e + 1]);
      pk1[e] = packbf2(p1[2 * e], p1[2 * e + 1]);
      x0[e]  = xhalf_u(pk0[e]);
      x1[e]  = xhalf_u(pk1[e]);
    }
    ABFrag bp;
#pragma unroll
    for (int e = 0; e < 4; ++e) {
      // hi==0: elems 0..7 = keys 0..7 (own st0), 8..15 = keys 8..15 (xchg st0)
      // hi==1: elems 0..7 = keys 16..23 (xchg st1), 8..15 = keys 24..31 (own st1)
      bp.u[e]     = hi ? x1[e] : pk0[e];
      bp.u[4 + e] = hi ? pk1[e] : x0[e];
    }

    // A fragments = vT rows (16 dq x 32 keys), two dq subtiles
    const unsigned short* vp0 = vT + ((size_t)h * kDQ + nl) * kN + kbase;
    const unsigned short* vp1 = vp0 + (size_t)16 * kN;
    ABFrag av0, av1;
    av0.q[0] = *(const uint4*)(vp0 + hi * 16);
    av0.q[1] = *(const uint4*)(vp0 + hi * 16 + 8);
    av1.q[0] = *(const uint4*)(vp1 + hi * 16);
    av1.q[1] = *(const uint4*)(vp1 + hi * 16 + 8);

    ot0 = wmma_bf16(av0.v, bp.v, ot0);
    ot1 = wmma_bf16(av1.v, bp.v, ot1);
  }

  // Store: lane = query qrow; ot0 -> dq 8*hi..+7, ot1 -> dq 16+8*hi..+7
  const float inv = 1.0f / lrun;
  float* orow = oBuf + (size_t)qrow * (kH * kDQ) + h * kDQ;
  alignas(16) float r[8];
#pragma unroll
  for (int j = 0; j < 8; ++j) r[j] = ot0[j] * inv;
  *(float4*)(orow + 8 * hi)     = *(const float4*)(r + 0);
  *(float4*)(orow + 8 * hi + 4) = *(const float4*)(r + 4);
#pragma unroll
  for (int j = 0; j < 8; ++j) r[j] = ot1[j] * inv;
  *(float4*)(orow + 16 + 8 * hi)     = *(const float4*)(r + 0);
  *(float4*)(orow + 16 + 8 * hi + 4) = *(const float4*)(r + 4);
}

// ---------------------------------------------------------------------------
extern "C" void kernel_launch(void* const* d_in, const int* /*in_sizes*/,
                              int /*n_in*/, void* d_out, int /*out_size*/,
                              void* d_ws, size_t /*ws_size*/,
                              hipStream_t stream) {
  const float* adj   = (const float*)d_in[0];
  const float* x     = (const float*)d_in[1];
  const float* tv    = (const float*)d_in[2];
  // d_in[3] = PNum (unused by reference math)
  const float* Wself = (const float*)d_in[4];
  const float* bself = (const float*)d_in[5];
  const float* Wn    = (const float*)d_in[6];
  const float* bn    = (const float*)d_in[7];
  const float* Wc    = (const float*)d_in[8];
  const float* bc    = (const float*)d_in[9];
  const float* Wq    = (const float*)d_in[10];
  const float* bq    = (const float*)d_in[11];
  const float* Wk    = (const float*)d_in[12];
  const float* bk    = (const float*)d_in[13];
  const float* Wv    = (const float*)d_in[14];
  const float* bv    = (const float*)d_in[15];
  const float* Wo    = (const float*)d_in[16];
  const float* bo    = (const float*)d_in[17];
  float* out = (float*)d_out;

  char* ws = (char*)d_ws;
  size_t off = 0;
  auto alloc = [&](size_t bytes) -> void* {
    void* p = ws + off;
    off += (bytes + 255) & ~(size_t)255;
    return p;
  };
  unsigned short* supTn = (unsigned short*)alloc((size_t)kH * kF * kN * 2);
  unsigned short* supTc = (unsigned short*)alloc((size_t)kH * kF * kN * 2);
  unsigned short* h_x   = (unsigned short*)alloc((size_t)kH * kN * kF * 2);
  unsigned short* h_n   = (unsigned short*)alloc((size_t)kH * kN * kF * 2);
  unsigned short* h_c   = (unsigned short*)alloc((size_t)kH * kN * kF * 2);
  unsigned short* qbuf  = (unsigned short*)alloc((size_t)kH * kN * kDQ * 2);
  unsigned short* kbuf  = (unsigned short*)alloc((size_t)kH * kN * kDQ * 2);
  unsigned short* vTbuf = (unsigned short*)alloc((size_t)kH * kDQ * kN * 2);
  float*          oBuf  = (float*)alloc((size_t)kN * kH * kDQ * 4);

  dim3 blk(256);
  dim3 gHN(kN / 128, kH);
  dim3 g1(kN / 128, 1);
  const float qscale = 0.17677669529663687f;  // 1/sqrt(DQ)

  // 1) h_x = relu(x @ Wself + bself)        bf16 row-major
  gemm_wmma_kernel<128, false, false, false, 1, true, 0><<<gHN, blk, 0, stream>>>(
      x, nullptr, 0L, kD, Wself, nullptr, (long)kD * kF, kF, bself, kF, nullptr,
      1.0f, nullptr, h_x, (long)kN * kF, kF, kD);

  // 2) supTn = ((x @ Wn) * t)^T             bf16 K-major
  gemm_wmma_kernel<128, false, false, false, 2, false, 1><<<gHN, blk, 0, stream>>>(
      x, nullptr, 0L, kD, Wn, nullptr, (long)kD * kF, kF, nullptr, 0, tv,
      1.0f, nullptr, supTn, (long)kF * kN, kN, kD);

  // 3) supTc = ((x @ Wc) * t)^T
  gemm_wmma_kernel<128, false, false, false, 2, false, 1><<<gHN, blk, 0, stream>>>(
      x, nullptr, 0L, kD, Wc, nullptr, (long)kD * kF, kF, nullptr, 0, tv,
      1.0f, nullptr, supTc, (long)kF * kN, kN, kD);

  // 4) h_n = relu(adj @ supn + bn)          K=4096 streaming pass over adj
  gemm_wmma_kernel<128, false, false, true, 1, true, 0><<<gHN, blk, 0, stream>>>(
      adj, nullptr, 0L, kN, nullptr, supTn, (long)kF * kN, kN, bn, kF, nullptr,
      1.0f, nullptr, h_n, (long)kN * kF, kF, kN);

  // 5) h_c = relu((adj + I) @ supc + bc)
  gemm_wmma_kernel<128, false, true, true, 1, true, 0><<<gHN, blk, 0, stream>>>(
      adj, nullptr, 0L, kN, nullptr, supTc, (long)kF * kN, kN, bc, kF, nullptr,
      1.0f, nullptr, h_c, (long)kN * kF, kF, kN);

  // 6) q = (h_x @ Wq + bq)*qscale ; k = h_n @ Wk + bk ; v (transposed)
  gemm_wmma_kernel<32, true, false, false, 3, true, 0><<<gHN, blk, 0, stream>>>(
      nullptr, h_x, (long)kN * kF, kF, Wq, nullptr, (long)kF * kDQ, kDQ, bq,
      kDQ, nullptr, qscale, nullptr, qbuf, (long)kN * kDQ, kDQ, kF);
  gemm_wmma_kernel<32, true, false, false, 0, true, 0><<<gHN, blk, 0, stream>>>(
      nullptr, h_n, (long)kN * kF, kF, Wk, nullptr, (long)kF * kDQ, kDQ, bk,
      kDQ, nullptr, 1.0f, nullptr, kbuf, (long)kN * kDQ, kDQ, kF);
  gemm_wmma_kernel<32, true, false, false, 0, true, 1><<<gHN, blk, 0, stream>>>(
      nullptr, h_c, (long)kN * kF, kF, Wv, nullptr, (long)kF * kDQ, kDQ, bv,
      kDQ, nullptr, 1.0f, nullptr, vTbuf, (long)kDQ * kN, kN, kF);

  // 7) flash attention -> oBuf [N][H*DQ] f32
  attn_kernel<<<gHN, blk, 0, stream>>>(qbuf, kbuf, vTbuf, oBuf);

  // 8) out = oBuf @ Wo + bo (f32)
  gemm_wmma_kernel<128, false, false, false, 0, true, 2><<<g1, blk, 0, stream>>>(
      oBuf, nullptr, 0L, kH * kDQ, Wo, nullptr, 0L, kF, bo, 0, nullptr,
      1.0f, out, nullptr, 0L, kF, kH * kDQ);
}
